// BalancedCELoss_77489799954598
// MI455X (gfx1250) — compile-verified, matrix operand order checked by reading, not run on hardware
//
#include <hip/hip_runtime.h>

// CDNA5 / gfx1250: wave32, WMMA 16x16x32 f16 -> f32
typedef __attribute__((ext_vector_type(16))) _Float16 v16h;
typedef __attribute__((ext_vector_type(8)))  float    v8f;

#define BLOCKS1   1024
#define THREADS1  256
#define WAVES_PB  (THREADS1 / 32)
#define CHUNK     512   // elements per wave per WMMA iteration (16 per lane)

// ln(2) folded into the class weights: log(x) = log2(x) * ln2 (v_log_f32 is log2)
#define W1LN2 (1.6f * 0.69314718055994530942f)
#define W0LN2 (0.4f * 0.69314718055994530942f)

__device__ __forceinline__ float waveReduceSum(float v) {
    v += __shfl_xor(v, 16, 32);
    v += __shfl_xor(v, 8, 32);
    v += __shfl_xor(v, 4, 32);
    v += __shfl_xor(v, 2, 32);
    v += __shfl_xor(v, 1, 32);
    return v;
}

// per-element weighted log term (returned as f16 for the WMMA accumulator)
// and accuracy update (exact f32 count).
__device__ __forceinline__ _Float16 perElem(float p, int t, float& corr) {
    bool  pos = (t == 1);
    float x   = pos ? p : 1.0f - p;                    // in (1e-6, 1)
    float per = (pos ? W1LN2 : W0LN2) * __builtin_amdgcn_logf(x); // v_log_f32
    corr += ((p >= 0.5f) == pos) ? 1.0f : 0.0f;        // round(p) == t
    return (_Float16)per;
}

__global__ __launch_bounds__(THREADS1)
void bce_stage1(const float* __restrict__ inp, const int* __restrict__ tgt,
                float* __restrict__ partials, int n) {
    const int lane       = threadIdx.x & 31;
    const int waveInBlk  = threadIdx.x >> 5;
    const int waveGlobal = blockIdx.x * WAVES_PB + waveInBlk;
    const int totalWaves = gridDim.x * WAVES_PB;
    const int nChunks    = n / CHUNK;

    // WMMA f32 accumulator and the all-ones B matrix (sum-over-K engine)
    v8f  c = {};
    v16h ones;
    #pragma unroll
    for (int i = 0; i < 16; ++i) ones[i] = (_Float16)1.0f;

    float corr      = 0.0f;   // exact integer count in f32
    float lossExtra = 0.0f;   // scalar tail contribution

    // Wave-uniform grid-stride over full 512-element chunks: EXEC stays all-1s.
    for (int chunk = waveGlobal; chunk < nChunks; chunk += totalWaves) {
        const float4* in4 = reinterpret_cast<const float4*>(inp) + (size_t)chunk * 128;
        const int4*   tg4 = reinterpret_cast<const int4*>(tgt)  + (size_t)chunk * 128;
        v16h a;
        #pragma unroll
        for (int j = 0; j < 4; ++j) {
            float4 p = in4[j * 32 + lane];   // global_load_b128, coalesced 2KB/wave
            int4   t = tg4[j * 32 + lane];
            a[j * 4 + 0] = perElem(p.x, t.x, corr);
            a[j * 4 + 1] = perElem(p.y, t.y, corr);
            a[j * 4 + 2] = perElem(p.z, t.z, corr);
            a[j * 4 + 3] = perElem(p.w, t.w, corr);
        }
        // D[m][n] = sum_k A[m][k] + C : one v_wmma per 512 elements, co-executes
        // with the VALU/trans stream (16-bit WMMA tracked as TRANS on CDNA5).
        c = __builtin_amdgcn_wmma_f32_16x16x32_f16(
                /*neg_a=*/false, a, /*neg_b=*/false, ones,
                /*c_mod=*/(short)0, c, /*reuse_a=*/false, /*reuse_b=*/false);
    }

    // Scalar tail (n % 512), handled by block 0 only.
    const int tailStart = nChunks * CHUNK;
    if (blockIdx.x == 0) {
        for (int i = tailStart + threadIdx.x; i < n; i += THREADS1) {
            float p = inp[i];
            int   t = tgt[i];
            bool  pos = (t == 1);
            float x   = pos ? p : 1.0f - p;
            lossExtra += (pos ? W1LN2 : W0LN2) * __builtin_amdgcn_logf(x);
            corr      += ((p >= 0.5f) == pos) ? 1.0f : 0.0f;
        }
    }

    // D has 16 identical columns -> sum of all 256 slots == 16 * true sum.
    float lanePart = (c[0] + c[1] + c[2] + c[3] + c[4] + c[5] + c[6] + c[7]) * 0.0625f
                   + lossExtra;

    float wl = waveReduceSum(lanePart);
    float wc = waveReduceSum(corr);

    __shared__ float sL[WAVES_PB], sC[WAVES_PB];
    if (lane == 0) { sL[waveInBlk] = wl; sC[waveInBlk] = wc; }
    __syncthreads();
    if (threadIdx.x == 0) {
        float bl = 0.0f, bc = 0.0f;
        #pragma unroll
        for (int w = 0; w < WAVES_PB; ++w) { bl += sL[w]; bc += sC[w]; }
        partials[2 * blockIdx.x + 0] = bl;
        partials[2 * blockIdx.x + 1] = bc;
    }
}

__global__ __launch_bounds__(256)
void bce_stage2(const float* __restrict__ partials, float* __restrict__ out,
                int nBlocks, float invN) {
    float l = 0.0f, cnt = 0.0f;
    for (int i = threadIdx.x; i < nBlocks; i += 256) {
        l   += partials[2 * i + 0];
        cnt += partials[2 * i + 1];
    }
    l   = waveReduceSum(l);
    cnt = waveReduceSum(cnt);

    __shared__ float sL[8], sC[8];
    const int lane = threadIdx.x & 31, w = threadIdx.x >> 5;
    if (lane == 0) { sL[w] = l; sC[w] = cnt; }
    __syncthreads();
    if (threadIdx.x == 0) {
        float tl = 0.0f, tc = 0.0f;
        #pragma unroll
        for (int i = 0; i < 8; ++i) { tl += sL[i]; tc += sC[i]; }
        out[0] = -tl * invN;   // loss = -sum/n
        out[1] =  tc * invN;   // acc  = correct/n
    }
}

extern "C" void kernel_launch(void* const* d_in, const int* in_sizes, int n_in,
                              void* d_out, int out_size, void* d_ws, size_t ws_size,
                              hipStream_t stream) {
    const float* inp = (const float*)d_in[0];
    const int*   tgt = (const int*)d_in[1];
    const int    n   = in_sizes[0];
    float* partials  = (float*)d_ws;          // 2 * BLOCKS1 floats (8 KB)
    float* out       = (float*)d_out;

    bce_stage1<<<BLOCKS1, THREADS1, 0, stream>>>(inp, tgt, partials, n);
    bce_stage2<<<1, 256, 0, stream>>>(partials, out, BLOCKS1, 1.0f / (float)n);
}
